// Decoder_66271345377668
// MI455X (gfx1250) — matrix operand order repaired and checked by reading, not compiled, not run on hardware
//
#include <hip/hip_runtime.h>

// Problem constants (from reference)
#define B_    8
#define S_    64
#define EMB_  768
#define H_    1024
#define D3_   128
#define CTX_  512
#define RWG_  16      // workgroups cooperating on the recurrence
#define RCOLS (H_ / RWG_)   // 64 output columns per workgroup

typedef __attribute__((ext_vector_type(16))) _Float16 v16h;
typedef __attribute__((ext_vector_type(8)))  _Float16 v8h;
typedef __attribute__((ext_vector_type(4)))  _Float16 v4h;
typedef __attribute__((ext_vector_type(8)))  float    v8f;
typedef __attribute__((ext_vector_type(4)))  unsigned v4u;
typedef __attribute__((ext_vector_type(8)))  unsigned v8u;

// ---------------------------------------------------------------------------
// CDNA5 tanh (v_tanh_f32) if the toolchain exposes it; guarded.
// ---------------------------------------------------------------------------
__device__ __forceinline__ float fast_tanh(float x) {
#if __has_builtin(__builtin_amdgcn_tanhf)
  return __builtin_amdgcn_tanhf(x);
#elif __has_builtin(__builtin_amdgcn_tanh_f32)
  return __builtin_amdgcn_tanh_f32(x);
#else
  return tanhf(x);
#endif
}

// ---------------------------------------------------------------------------
// WMMA fragment loads, all-f16 sources, contiguous 16B accesses.
// A 16x32 (MxK): lanes 0-15 M=lane, K k0..k0+15
//                lanes16-31 M=l-16, K k0+16..k0+31 (matches loadB_h)
// ---------------------------------------------------------------------------
__device__ __forceinline__ v16h loadA_h(const _Float16* rowp, int k0, int lane) {
  const _Float16* p = rowp + k0 + ((lane >> 4) << 4);
  v8h a0 = *(const v8h*)p;          // 16B
  v8h a1 = *(const v8h*)(p + 8);    // 16B
  return __builtin_shufflevector(a0, a1, 0, 1, 2, 3, 4, 5, 6, 7,
                                 8, 9, 10, 11, 12, 13, 14, 15);
}

// B 32x16 (KxN) from an N-major (transposed, N x K) f16 image:
// N = n0+(lane&15); lanes 0-15 hold K k0..k0+15, lanes 16-31 K k0+16..k0+31.
__device__ __forceinline__ v16h loadB_h(const _Float16* BT, int ldk, int n0,
                                        int k0, int lane) {
  const _Float16* p = BT + (n0 + (lane & 15)) * ldk + k0 + ((lane >> 4) << 4);
  v8h b0 = *(const v8h*)p;          // 32B contiguous per lane
  v8h b1 = *(const v8h*)(p + 8);
  return __builtin_shufflevector(b0, b1, 0, 1, 2, 3, 4, 5, 6, 7,
                                 8, 9, 10, 11, 12, 13, 14, 15);
}
// C/D 16x16 f32: VGPR r -> M = r + 8*(lane>=16), N = lane&15

#define WMMA(acc, a, b)                                                        \
  acc = __builtin_amdgcn_wmma_f32_16x16x32_f16(false, a, false, b, (short)0,   \
                                               acc, false, false)

// ---------------------------------------------------------------------------
// One-time staging kernels (f32 -> f16, transpose K x N -> N x K).
// ---------------------------------------------------------------------------
__global__ void k_cvt(const float* __restrict__ src, _Float16* __restrict__ dst,
                      int n4) {                      // n4 = elements/4
  for (int i = blockIdx.x * blockDim.x + threadIdx.x; i < n4;
       i += gridDim.x * blockDim.x) {
    float4 x = ((const float4*)src)[i];
    v4h h = {(_Float16)x.x, (_Float16)x.y, (_Float16)x.z, (_Float16)x.w};
    ((v4h*)dst)[i] = h;
  }
}

__global__ void k_transpose(const float* __restrict__ src,
                            _Float16* __restrict__ dst, int K, int N) {
  int total = K * N;
  for (int i = blockIdx.x * blockDim.x + threadIdx.x; i < total;
       i += gridDim.x * blockDim.x) {
    int n = i / K, k = i - n * K;
    dst[i] = (_Float16)src[k * N + n];               // dst[n][k] = src[k][n]
  }
}

__global__ void k_gather_cvt(const int* __restrict__ ids,
                             const float* __restrict__ emb,
                             _Float16* __restrict__ feats) {
  int total = B_ * S_ * EMB_;
  for (int i = blockIdx.x * blockDim.x + threadIdx.x; i < total;
       i += gridDim.x * blockDim.x) {
    int m = i / EMB_, k = i - m * EMB_;
    feats[i] = (_Float16)emb[ids[m] * EMB_ + k];
  }
}

// h0 = context[:, CTX-1, :] as f16; also resets the inter-WG barrier counter
// (every launch, so graph replays stay deterministic).
__global__ void k_h0(const float* __restrict__ ctx, _Float16* __restrict__ h0h,
                     unsigned* __restrict__ cnt) {
  int i = blockIdx.x * blockDim.x + threadIdx.x;
  if (i == 0) *cnt = 0u;
  if (i < B_ * H_) {
    int b = i >> 10, j = i & (H_ - 1);
    h0h[i] = (_Float16)ctx[(b * CTX_ + (CTX_ - 1)) * H_ + j];
  }
}

// ---------------------------------------------------------------------------
// K1: ctx_proj = context(4096x1024,f16) @ W_att_h + b  -> f32 (tanh input)
// ---------------------------------------------------------------------------
__global__ void k_ctxproj(const _Float16* __restrict__ A,
                          const _Float16* __restrict__ BT,
                          const float* __restrict__ bias,
                          float* __restrict__ out) {
  int lane = threadIdx.x & 31;
  int tile = blockIdx.x * (blockDim.x >> 5) + (threadIdx.x >> 5);  // 2048 tiles
  int m0 = (tile >> 3) * 16, n0 = (tile & 7) * 16;
  const _Float16* rowp = A + (m0 + (lane & 15)) * H_;
  v8f acc = {};
#pragma unroll 4
  for (int k0 = 0; k0 < H_; k0 += 32)
    WMMA(acc, loadA_h(rowp, k0, lane), loadB_h(BT, H_, n0, k0, lane));
  int n = n0 + (lane & 15);
  float bv = bias[n];
  int mh = (lane >> 4) * 8;
#pragma unroll
  for (int r = 0; r < 8; ++r)
    out[(m0 + mh + r) * D3_ + n] = acc[r] + bv;
}

// ---------------------------------------------------------------------------
// K2: U = feats(512x768,f16 pre-gathered) @ W_in + b_in -> f16
// ---------------------------------------------------------------------------
__global__ void k_embed(const _Float16* __restrict__ feats,
                        const _Float16* __restrict__ BT,
                        const float* __restrict__ bias,
                        _Float16* __restrict__ U) {
  int lane = threadIdx.x & 31;
  int tile = blockIdx.x * (blockDim.x >> 5) + (threadIdx.x >> 5);  // 2048 tiles
  int m0 = (tile >> 6) * 16, n0 = (tile & 63) * 16;
  const _Float16* rowp = feats + (m0 + (lane & 15)) * EMB_;
  v8f acc = {};
#pragma unroll 4
  for (int k0 = 0; k0 < EMB_; k0 += 32)
    WMMA(acc, loadA_h(rowp, k0, lane), loadB_h(BT, EMB_, n0, k0, lane));
  int n = n0 + (lane & 15);
  float bv = bias[n];
  int mh = (lane >> 4) * 8;
#pragma unroll
  for (int r = 0; r < 8; ++r)
    U[(m0 + mh + r) * H_ + n] = (_Float16)(acc[r] + bv);
}

// ---------------------------------------------------------------------------
// K3: serial recurrence across 16 cooperating workgroups.
//     WG g owns output columns [g*64, g*64+64).  Its 128KB B-panel of WsT is
//     step-invariant -> staged ONCE into LDS by the Tensor Data Mover
//     (2D D# tile: 1024 x 64 x 2B, stride 1024; ISA 8.3/8.4), so per-step
//     global traffic is only the 16KB h slab.  Step sync = cluster barrier
//     (real if dispatched as a cluster, architectural NOP otherwise) backed
//     by a device-scope atomic arrival counter.
// ---------------------------------------------------------------------------
__global__ void k_recur(const _Float16* __restrict__ WsT,
                        const float* __restrict__ bs,
                        const _Float16* __restrict__ h0h,
                        _Float16* __restrict__ Hs,
                        unsigned* __restrict__ cnt) {
  extern __shared__ _Float16 bpan[];                 // [RCOLS][H_] = 128 KB
  int tid = threadIdx.x, lane = tid & 31, wave = tid >> 5;   // 4 waves
  int blk = blockIdx.x;

  // ---- TDM: stage B-panel (WsT rows blk*64 .. +63) into LDS ---------------
  if (tid < 32) {
    unsigned long long ga =
        (unsigned long long)(const void*)(WsT + (size_t)blk * RCOLS * H_);
    v4u g0;
    g0[0] = 1u;                                    // count=1, user descriptor
    g0[1] = 0u;                                    // lds_addr = dynamic base
    g0[2] = (unsigned)ga;                          // global_addr[31:0]
    g0[3] = (unsigned)((ga >> 32) & 0x01ffffffu) | 0x80000000u;  // type=2
    v8u g1;
    g1[0] = 1u << 16;                              // data_size=1 (2 bytes)
    g1[1] = (unsigned)H_ << 16;                    // tensor_dim0[15:0]=1024
    g1[2] = (unsigned)RCOLS << 16;                 // tensor_dim1[15:0]=64
    g1[3] = (unsigned)H_ << 16;                    // tile_dim0=1024
    g1[4] = (unsigned)RCOLS;                       // tile_dim1=64, tile_dim2=0
    g1[5] = (unsigned)H_;                          // tensor_dim0_stride=1024
    g1[6] = 0u;
    g1[7] = 0u;
    asm volatile("tensor_load_to_lds %0, %1" :: "s"(g0), "s"(g1) : "memory");
    __builtin_amdgcn_s_wait_tensorcnt(0);
  }
  __syncthreads();                                 // panel visible to all waves

  int arow = (lane & 15) < B_ ? (lane & 15) : 0;   // clamp padded rows (M=8)
  int n0l = wave * 16;                             // local n-tile in panel
  int ng = blk * RCOLS + n0l + (lane & 15);        // global column
  float bv = bs[ng];
  int mh = (lane >> 4) * 8;

  for (int t = 0; t < S_; ++t) {
    const _Float16* rowp =
        (t == 0 ? h0h : Hs + (t - 1) * (B_ * H_)) + arow * H_;
    v8f acc = {};
#pragma unroll 4
    for (int k0 = 0; k0 < H_; k0 += 32)
      WMMA(acc, loadA_h(rowp, k0, lane), loadB_h(bpan, H_, n0l, k0, lane));
#pragma unroll
    for (int r = 0; r < 8; ++r) {
      int m = mh + r;
      if (m < B_)
        Hs[(t * B_ + m) * H_ + ng] = (_Float16)(acc[r] + bv);
    }
    // ---- step barrier across the 16 workgroups ---------------------------
    __threadfence();                               // flush h-slab to device
    __syncthreads();
    if (tid == 0) {
      __builtin_amdgcn_s_cluster_barrier();        // s_barrier_signal/wait -3
      __hip_atomic_fetch_add(cnt, 1u, __ATOMIC_ACQ_REL,
                             __HIP_MEMORY_SCOPE_AGENT);
      unsigned target = (unsigned)(RWG_ * (t + 1));
      while (__hip_atomic_load(cnt, __ATOMIC_ACQUIRE,
                               __HIP_MEMORY_SCOPE_AGENT) < target)
        __builtin_amdgcn_s_sleep(1);
    }
    __syncthreads();
  }
}

// ---------------------------------------------------------------------------
// K4: Q[(t,b),d] = U[(b,t),:]@Wa[0:H] + Hs[(t,b),:]@Wa[H:2H] + b_att_in -> f32
// ---------------------------------------------------------------------------
__global__ void k_att(const _Float16* __restrict__ U,
                      const _Float16* __restrict__ Hs,
                      const _Float16* __restrict__ WaT,
                      const float* __restrict__ bias, float* __restrict__ Q) {
  int lane = threadIdx.x & 31;
  int tile = blockIdx.x * (blockDim.x >> 5) + (threadIdx.x >> 5);  // 256 tiles
  int m0 = (tile >> 3) * 16, n0 = (tile & 7) * 16;
  int r = m0 + (lane & 15);                          // row == t*B + b
  int t = r >> 3, b = r & 7;
  const _Float16* rowU = U + (b * S_ + t) * H_;
  const _Float16* rowH = Hs + r * H_;
  v8f acc = {};
#pragma unroll 4
  for (int k0 = 0; k0 < H_; k0 += 32)
    WMMA(acc, loadA_h(rowU, k0, lane), loadB_h(WaT, 2 * H_, n0, k0, lane));
#pragma unroll 4
  for (int k0 = 0; k0 < H_; k0 += 32)
    WMMA(acc, loadA_h(rowH, k0, lane),
         loadB_h(WaT, 2 * H_, n0, H_ + k0, lane));
  int n = n0 + (lane & 15);
  float bv = bias[n];
  int mh = (lane >> 4) * 8;
#pragma unroll
  for (int rr = 0; rr < 8; ++rr)
    Q[(m0 + mh + rr) * D3_ + n] = acc[rr] + bv;
}

// ---------------------------------------------------------------------------
// K5: atts[b,t,c] = sum_d V[d] * tanh(Q[(t,b),d] + ctx_proj[(b,c),d])
//     33.5M transcendentals; ctx_proj (2MB) stays L2-resident across S=64
//     reuses.  Block per (b,t).
// ---------------------------------------------------------------------------
__global__ void k_tanh_att(const float* __restrict__ Q,
                           const float* __restrict__ ctxp,
                           const float* __restrict__ V,
                           float* __restrict__ out) {
  __shared__ float q[D3_];
  __shared__ float vv[D3_];
  int b = blockIdx.x >> 6, t = blockIdx.x & 63;
  int r = t * B_ + b;
  for (int i = threadIdx.x; i < D3_; i += blockDim.x) {
    q[i] = Q[r * D3_ + i];
    vv[i] = V[i];
  }
  __syncthreads();
  for (int c = threadIdx.x; c < CTX_; c += blockDim.x) {
    const float* cp = ctxp + (b * CTX_ + c) * D3_;
    if (c + (int)blockDim.x < CTX_)                      // stream-ahead hint
      __builtin_prefetch(cp + blockDim.x * D3_, 0, 0);   // global_prefetch_b8
    float acc = 0.f;
#pragma unroll 8
    for (int d = 0; d < D3_; d += 4) {
      float4 x = *(const float4*)(cp + d);
      acc += vv[d + 0] * fast_tanh(q[d + 0] + x.x);
      acc += vv[d + 1] * fast_tanh(q[d + 1] + x.y);
      acc += vv[d + 2] * fast_tanh(q[d + 2] + x.z);
      acc += vv[d + 3] * fast_tanh(q[d + 3] + x.w);
    }
    out[(b * S_ + t) * CTX_ + c] = acc;
  }
}

// prts: softmax over a size-1 axis is all-ones -> argmax == 0 everywhere.
__global__ void k_prts(float* __restrict__ out) {
  int i = blockIdx.x * blockDim.x + threadIdx.x;
  if (i < S_ * B_) out[B_ * S_ * CTX_ + i] = 0.f;    // bit pattern 0 == int 0
}

// ---------------------------------------------------------------------------
extern "C" void kernel_launch(void* const* d_in, const int* in_sizes, int n_in,
                              void* d_out, int out_size, void* d_ws,
                              size_t ws_size, hipStream_t stream) {
  const int*   ids      = (const int*)d_in[0];
  // d_in[1] attn_mask, d_in[2] seq_type_id, d_in[3] h_first: unused by ref
  const float* ctx      = (const float*)d_in[4];
  const float* emb      = (const float*)d_in[5];
  const float* W_in     = (const float*)d_in[6];
  const float* b_in     = (const float*)d_in[7];
  const float* W_s      = (const float*)d_in[8];
  const float* b_s      = (const float*)d_in[9];
  const float* W_att_in = (const float*)d_in[10];
  const float* b_att_in = (const float*)d_in[11];
  const float* W_att_h  = (const float*)d_in[12];
  const float* b_att_h  = (const float*)d_in[13];
  const float* V        = (const float*)d_in[14];
  float* out = (float*)d_out;

  // ---- workspace carve-up -------------------------------------------------
  float* wsf  = (float*)d_ws;
  float* ctxp = wsf;                           // 8*512*128       = 524288 f32
  float* Q    = wsf + 524288;                  // 512*128         =  65536 f32
  unsigned* cnt = (unsigned*)(wsf + 589824);   // barrier counter (16f pad)
  _Float16* h = (_Float16*)(wsf + 589840);
  _Float16* ctxh   = h;                        // 4096*1024  = 4194304 h
  _Float16* feats  = h + 4194304;              // 512*768    =  393216 h
  _Float16* Uh     = h + 4587520;              // 512*1024   =  524288 h
  _Float16* Hsh    = h + 5111808;              // 512*1024   =  524288 h
  _Float16* WahT   = h + 5636096;              // 128x1024   =  131072 h
  _Float16* WinT   = h + 5767168;              // 1024x768   =  786432 h
  _Float16* WsT    = h + 6553600;              // 1024x1024  = 1048576 h
  _Float16* WaT    = h + 7602176;              // 128x2048   =  262144 h
  _Float16* h0h    = h + 7864320;              // 8x1024     =    8192 h
  // total: ~2.4 MB f32 + ~15.8 MB f16

  // ---- stage f16 operand images (one-time, ~6.8M elements total) ----------
  k_cvt       <<<1024, 256, 0, stream>>>(ctx, ctxh, (B_ * CTX_ * H_) / 4);
  k_gather_cvt<<< 384, 256, 0, stream>>>(ids, emb, feats);
  k_transpose <<< 128, 256, 0, stream>>>(W_att_h, WahT, H_, D3_);
  k_transpose <<< 768, 256, 0, stream>>>(W_in, WinT, EMB_, H_);
  k_transpose <<<1024, 256, 0, stream>>>(W_s, WsT, H_, H_);
  k_transpose <<< 256, 256, 0, stream>>>(W_att_in, WaT, 2 * H_, D3_);
  k_h0        <<<  32, 256, 0, stream>>>(ctx, h0h, cnt);

  // ---- compute ------------------------------------------------------------
  k_ctxproj <<<256, 256, 0, stream>>>(ctxh, WahT, b_att_h, ctxp);  // 2048 tiles
  k_embed   <<<256, 256, 0, stream>>>(feats, WinT, b_in, Uh);      // 2048 tiles
  k_recur   <<<RWG_, 128, RCOLS * H_ * sizeof(_Float16), stream>>>(
      WsT, b_s, h0h, Hsh, cnt);
  k_att     <<<32, 256, 0, stream>>>(Uh, Hsh, WaT, b_att_in, Q);   // 256 tiles
  k_tanh_att<<<B_ * S_, 256, 0, stream>>>(Q, ctxp, V, out);
  k_prts    <<<2, 256, 0, stream>>>(out);
}
